// BFANet_mink_63840393888176
// MI455X (gfx1250) — matrix operand order repaired and checked by reading, not compiled, not run on hardware
//
#include <hip/hip_runtime.h>

// ---------------------------------------------------------------------------
// Types for CDNA5 WMMA (wave32): bf16 A/B fragments (16 bf16/lane = 8 VGPRs),
// f32 C/D accumulator (8 f32/lane = 8 VGPRs).
// ---------------------------------------------------------------------------
typedef __attribute__((ext_vector_type(16))) __bf16         v16bf;
typedef __attribute__((ext_vector_type(8)))  float          v8f;
typedef __attribute__((ext_vector_type(8)))  unsigned short u16x8;

#define MT 4  // voxel (M) 16-row tiles per wave: B fragment reused by 4 WMMAs

__device__ __forceinline__ unsigned short f2bf(float f) {
  // round-to-nearest-even f32 -> bf16 (bit pattern)
  unsigned int u = __float_as_uint(f);
  unsigned int r = u + 0x7FFFu + ((u >> 16) & 1u);
  return (unsigned short)(r >> 16);
}

// ---------------------------------------------------------------------------
// Pack weights [KK, Cin, Cout] (f32) into WMMA-B fragment blocks (bf16).
// Block index = (kk*nCC + cc)*nCT + ct, each block = 32 lanes x 16 elems.
// Lane L, elem e holds B[K = cc*32 + (e&7) + 8*(L>=16) + 16*(e>=8)][N = ct*16 + (L&15)],
// zero-padded outside [Cin, Cout].
// ---------------------------------------------------------------------------
__global__ void pack_w_kernel(const float* __restrict__ W, int KK, int Cin, int Cout,
                              unsigned short* __restrict__ dst, int total) {
  int nCT = (Cout + 15) >> 4;
  int nCC = (Cin + 31) >> 5;
  for (int idx = blockIdx.x * blockDim.x + threadIdx.x; idx < total;
       idx += gridDim.x * blockDim.x) {
    int e    = idx & 15;
    int lane = (idx >> 4) & 31;
    int blk  = idx >> 9;
    int ct   = blk % nCT;
    int t2   = blk / nCT;
    int cc   = t2 % nCC;
    int kk   = t2 / nCC;
    int K    = (e & 7) + ((lane >> 4) << 3) + ((e >> 3) << 4);
    int ci   = (cc << 5) + K;
    int co   = (ct << 4) + (lane & 15);
    float v  = 0.f;
    if (ci < Cin && co < Cout) v = W[((size_t)kk * Cin + ci) * Cout + co];
    dst[idx] = f2bf(v);
  }
}

// Convert layer-0 features [N, Cin] f32 -> [N, cinP] bf16 (zero padded)
__global__ void pack_feat_kernel(const float* __restrict__ x, int N, int Cin, int cinP,
                                 unsigned short* __restrict__ out) {
  int total = N * cinP;
  for (int idx = blockIdx.x * blockDim.x + threadIdx.x; idx < total;
       idx += gridDim.x * blockDim.x) {
    int n = idx / cinP;
    int c = idx - n * cinP;
    out[idx] = (c < Cin) ? f2bf(x[(size_t)n * Cin + c]) : (unsigned short)0;
  }
}

// ---------------------------------------------------------------------------
// Sparse-conv / dense GEMM via WMMA bf16 -> f32.
//   y[m, co] = sum_k sum_ci gat(m,k,ci) * Wpack(k,ci,co)   (+ optional bias)
// gat(m,k,ci) = (nbr[m*KK+k] >= 0) ? xb[nbr][ci] : 0 ; dense mode: nbr==NULL.
// One wave (block of 32) computes MT stacked 16x16 output tiles for one
// 16-wide Cout tile (blockIdx.y), so each weight (B) fragment load feeds MT
// back-to-back v_wmma ops.
// A fragment per ISA 16-bit A 16x32 layout: lane L handles row M = L&15,
// K-chunks {kOff..kOff+7} and {kOff+16..kOff+23}, kOff = 8*(L>=16) -> two
// aligned 16-byte loads.
// ---------------------------------------------------------------------------
__global__ void sconv_wmma_kernel(const unsigned short* __restrict__ xb, int cinP,
                                  const int* __restrict__ nbr, int KK,
                                  const unsigned short* __restrict__ wp,
                                  const float* __restrict__ bias, int coutA,
                                  float* __restrict__ y, int coutP, int nvox) {
  const int nCC  = cinP >> 5;
  const int nCT  = coutP >> 4;
  const int ct   = blockIdx.y;        // Cout tile (no integer division needed)
  const int mt0  = blockIdx.x * MT;   // first voxel 16-row tile
  const int lane = threadIdx.x;
  const bool hiK  = lane >= 16;
  const int  kOff = hiK ? 8 : 0;

  int rowA[MT];
#pragma unroll
  for (int i = 0; i < MT; ++i) rowA[i] = (mt0 + i) * 16 + (lane & 15);

  const v8f vzero = {};
  v8f acc[MT];
#pragma unroll
  for (int i = 0; i < MT; ++i) acc[i] = vzero;

  // weight fragment base for this wave's Cout tile; stride between
  // consecutive (k*nCC+cc) blocks is nCT*512 bf16 elements
  const unsigned short* wpBase = wp + ((size_t)ct << 9) + (lane << 4);
  const size_t blkStride = (size_t)nCT << 9;

  for (int k = 0; k < KK; ++k) {
    int vIdx[MT];
#pragma unroll
    for (int i = 0; i < MT; ++i) {
      int r = rowA[i];
      vIdx[i] = nbr ? ((r < nvox) ? nbr[(size_t)r * KK + k] : -1)
                    : ((r < nvox) ? r : -1);
    }
    for (int cc = 0; cc < nCC; ++cc) {
      const v16bf B = *(const v16bf*)(wpBase + (size_t)(k * nCC + cc) * blkStride);
      if (k + 1 < KK)  // prefetch next tap's weight fragment (global_prefetch_b8)
        __builtin_prefetch(wpBase + (size_t)((k + 1) * nCC + cc) * blkStride, 0, 0);
#pragma unroll
      for (int i = 0; i < MT; ++i) {
        u16x8 lo = {}, hi = {};
        int v = vIdx[i];
        if (v >= 0) {
          const unsigned short* s = xb + (size_t)v * cinP + (cc << 5) + kOff;
          lo = *(const u16x8*)(s);
          hi = *(const u16x8*)(s + 16);
        }
        union { v16bf v16; u16x8 h[2]; } A;
        A.h[0] = lo;
        A.h[1] = hi;
        acc[i] = __builtin_amdgcn_wmma_f32_16x16x32_bf16(false, A.v16, false, B,
                                                         (short)0, acc[i], false, false);
      }
    }
  }

  // C/D layout: lane L holds column N = L&15, rows M = r + 8*(L>=16)
  int col = ct * 16 + (lane & 15);
  float add = (bias != nullptr && col < coutA) ? bias[col] : 0.f;
#pragma unroll
  for (int i = 0; i < MT; ++i) {
    int rbase = (mt0 + i) * 16 + (hiK ? 8 : 0);
#pragma unroll
    for (int r = 0; r < 8; ++r) {
      int row = rbase + r;
      if (row < nvox) y[(size_t)row * coutP + col] = acc[i][r] + add;
    }
  }
}

// ---------------------------------------------------------------------------
// BN batch statistics: per-channel sum & sumsq (C is a power of two: 32/64).
// ---------------------------------------------------------------------------
__global__ void zero_buf_kernel(float* __restrict__ p, int n) {
  int i = blockIdx.x * blockDim.x + threadIdx.x;
  if (i < n) p[i] = 0.f;
}

__global__ void bn_stats_kernel(const float* __restrict__ y, int coutP, int C, int log2C,
                                int N, float* __restrict__ stats) {
  __shared__ float s[128];
  int t = threadIdx.x;
  if (t < 2 * C) s[t] = 0.f;
  __syncthreads();
  int total = N << log2C;
  for (int idx = blockIdx.x * blockDim.x + t; idx < total; idx += gridDim.x * blockDim.x) {
    int c   = idx & (C - 1);
    int n   = idx >> log2C;
    float v = y[(size_t)n * coutP + c];
    atomicAdd(&s[c], v);
    atomicAdd(&s[C + c], v * v);
  }
  __syncthreads();
  if (t < 2 * C) atomicAdd(&stats[t], s[t]);
}

// BN apply + activation (ReLU, or PReLU if alpha != NULL) + bf16 repack
__global__ void bn_apply_kernel(const float* __restrict__ y, int coutP, int C, int N,
                                const float* __restrict__ stats,
                                const float* __restrict__ g, const float* __restrict__ b,
                                const float* __restrict__ alpha,
                                unsigned short* __restrict__ xout, int cinPnext) {
  int total = N * cinPnext;
  for (int idx = blockIdx.x * blockDim.x + threadIdx.x; idx < total;
       idx += gridDim.x * blockDim.x) {
    int n = idx / cinPnext;
    int c = idx - n * cinPnext;
    unsigned short o = 0;
    if (c < C) {
      float invN = 1.0f / (float)N;
      float mean = stats[c] * invN;
      float var  = stats[C + c] * invN - mean * mean;
      float h = (y[(size_t)n * coutP + c] - mean) * rsqrtf(var + 1e-5f) * g[c] + b[c];
      if (alpha) h = (h >= 0.f) ? h : alpha[0] * h;
      else       h = fmaxf(h, 0.f);
      o = f2bf(h);
    }
    xout[idx] = o;
  }
}

// ---------------------------------------------------------------------------
// Final voxel -> point gather, sigmoid on margin, 4-way concat output:
// [sem_p (P,20) | mar_p (P,1) | sem_p (P,20) | mar_p (P,1)]
// ---------------------------------------------------------------------------
__global__ void gather_out_kernel(const float* __restrict__ ySem,  // [N,32], 20 valid
                                  const float* __restrict__ yMar,  // [N,16], 1 valid
                                  const long long* __restrict__ v2p, int npts,
                                  float* __restrict__ out) {
  int total = npts * 21;
  size_t o1 = (size_t)npts * 20;
  size_t o2 = (size_t)npts * 21;
  size_t o3 = (size_t)npts * 41;
  for (int idx = blockIdx.x * blockDim.x + threadIdx.x; idx < total;
       idx += gridDim.x * blockDim.x) {
    int p = idx / 21;
    int j = idx - p * 21;
    int v = (int)v2p[p];
    if (j < 20) {
      float s = ySem[(size_t)v * 32 + j];
      out[(size_t)p * 20 + j]      = s;
      out[o2 + (size_t)p * 20 + j] = s;
    } else {
      float m = 1.0f / (1.0f + expf(-yMar[(size_t)v * 16]));
      out[o1 + p] = m;
      out[o3 + p] = m;
    }
  }
}

// ---------------------------------------------------------------------------
extern "C" void kernel_launch(void* const* d_in, const int* in_sizes, int n_in,
                              void* d_out, int out_size, void* d_ws, size_t ws_size,
                              hipStream_t stream) {
  const float*     feat = (const float*)d_in[0];
  const long long* v2p  = (const long long*)d_in[2];
  const int*       nbr  = (const int*)d_in[3];
  const float* W0 = (const float*)d_in[4],  *g0 = (const float*)d_in[5],  *b0 = (const float*)d_in[6];
  const float* W1 = (const float*)d_in[7],  *g1 = (const float*)d_in[8],  *b1 = (const float*)d_in[9];
  const float* W2 = (const float*)d_in[10], *g2 = (const float*)d_in[11], *b2 = (const float*)d_in[12];
  const float* W3 = (const float*)d_in[13], *g3 = (const float*)d_in[14], *b3 = (const float*)d_in[15];
  const float* semW1 = (const float*)d_in[16], *semG = (const float*)d_in[17], *semB = (const float*)d_in[18];
  const float* semA  = (const float*)d_in[19];
  const float* semW2 = (const float*)d_in[20], *semB2 = (const float*)d_in[21];
  const float* marW1 = (const float*)d_in[22], *marG = (const float*)d_in[23], *marB = (const float*)d_in[24];
  const float* marA  = (const float*)d_in[25];
  const float* marW2 = (const float*)d_in[26], *marB2 = (const float*)d_in[27];

  const int N  = in_sizes[0] / 6;   // 40000
  const int NP = in_sizes[2];       // 120000
  float* out = (float*)d_out;

  // ---- workspace layout ----
  char* ws = (char*)d_ws;
  size_t off = 0;
  auto take = [&](size_t bytes) -> char* {
    char* p = ws + off;
    off = (off + bytes + 255) & ~(size_t)255;
    return p;
  };
  unsigned short* xbA  = (unsigned short*)take((size_t)N * 64 * 2);
  unsigned short* xbB  = (unsigned short*)take((size_t)N * 64 * 2);
  float*          ybuf = (float*)take((size_t)N * 64 * 4);
  float*          ySem = (float*)take((size_t)N * 32 * 4);
  float*          yMar = (float*)take((size_t)N * 16 * 4);
  float*          stats = (float*)take(512);
  unsigned short* wp0  = (unsigned short*)take((size_t)27 * 2 * 512 * 2);
  unsigned short* wp1  = (unsigned short*)take((size_t)27 * 2 * 512 * 2);
  unsigned short* wp2  = (unsigned short*)take((size_t)27 * 4 * 512 * 2);
  unsigned short* wp3  = (unsigned short*)take((size_t)27 * 2 * 2 * 512 * 2);
  unsigned short* wpS1 = (unsigned short*)take((size_t)2 * 512 * 2);
  unsigned short* wpS2 = (unsigned short*)take((size_t)2 * 512 * 2);
  unsigned short* wpM1 = (unsigned short*)take((size_t)2 * 512 * 2);
  unsigned short* wpM2 = (unsigned short*)take((size_t)1 * 512 * 2);

  const int NM16 = (N + 15) / 16;           // 16-row tiles
  const int NMB  = (NM16 + MT - 1) / MT;    // blocks of MT tiles

  auto packW = [&](const float* W, int KK, int Cin, int Cout, unsigned short* dst) {
    int nCT = (Cout + 15) >> 4, nCC = (Cin + 31) >> 5;
    int total = KK * nCC * nCT * 512;
    pack_w_kernel<<<(total + 255) / 256, 256, 0, stream>>>(W, KK, Cin, Cout, dst, total);
  };
  auto conv = [&](const unsigned short* xin, int cinP, const int* nb, int KK,
                  const unsigned short* wp, const float* bias, int coutA,
                  float* yo, int coutP) {
    dim3 grid(NMB, coutP >> 4);
    sconv_wmma_kernel<<<grid, 32, 0, stream>>>(xin, cinP, nb, KK, wp, bias, coutA,
                                               yo, coutP, N);
  };
  auto bnStage = [&](const float* yin, int coutP, int C, int log2C, const float* g,
                     const float* b, const float* alpha, unsigned short* xo, int cinPn) {
    zero_buf_kernel<<<1, 128, 0, stream>>>(stats, 2 * C);
    bn_stats_kernel<<<512, 256, 0, stream>>>(yin, coutP, C, log2C, N, stats);
    int total = N * cinPn;
    bn_apply_kernel<<<(total + 255) / 256, 256, 0, stream>>>(yin, coutP, C, N, stats, g, b,
                                                             alpha, xo, cinPn);
  };

  // ---- pack weights (tiny) ----
  packW(W0, 27, 6, 32, wp0);
  packW(W1, 27, 32, 32, wp1);
  packW(W2, 27, 32, 64, wp2);
  packW(W3, 27, 64, 32, wp3);
  packW(semW1, 1, 32, 32, wpS1);
  packW(semW2, 1, 32, 20, wpS2);
  packW(marW1, 1, 32, 32, wpM1);
  packW(marW2, 1, 32, 1, wpM2);

  // ---- layer-0 input pack ----
  {
    int total = N * 32;
    pack_feat_kernel<<<(total + 255) / 256, 256, 0, stream>>>(feat, N, 6, 32, xbA);
  }

  // ---- 4 sparse conv + BN + ReLU layers (ping-pong xbA/xbB) ----
  conv(xbA, 32, nbr, 27, wp0, nullptr, 32, ybuf, 32);
  bnStage(ybuf, 32, 32, 5, g0, b0, nullptr, xbB, 32);
  conv(xbB, 32, nbr, 27, wp1, nullptr, 32, ybuf, 32);
  bnStage(ybuf, 32, 32, 5, g1, b1, nullptr, xbA, 32);
  conv(xbA, 32, nbr, 27, wp2, nullptr, 64, ybuf, 64);
  bnStage(ybuf, 64, 64, 6, g2, b2, nullptr, xbB, 64);
  conv(xbB, 64, nbr, 27, wp3, nullptr, 32, ybuf, 32);
  bnStage(ybuf, 32, 32, 5, g3, b3, nullptr, xbA, 32);  // xbA = final features (bf16)

  // ---- semantic head: Linear -> BN -> PReLU -> Linear(+bias) ----
  conv(xbA, 32, nullptr, 1, wpS1, nullptr, 32, ybuf, 32);
  bnStage(ybuf, 32, 32, 5, semG, semB, semA, xbB, 32);
  conv(xbB, 32, nullptr, 1, wpS2, semB2, 20, ySem, 32);

  // ---- margin head: Linear -> BN -> PReLU -> Linear(+bias) (sigmoid later) ----
  conv(xbA, 32, nullptr, 1, wpM1, nullptr, 32, ybuf, 32);
  bnStage(ybuf, 32, 32, 5, marG, marB, marA, xbB, 32);
  conv(xbB, 32, nullptr, 1, wpM2, marB2, 1, yMar, 16);

  // ---- voxel -> point gather + sigmoid + 4-way concat ----
  {
    int total = NP * 21;
    gather_out_kernel<<<(total + 255) / 256, 256, 0, stream>>>(ySem, yMar, v2p, NP, out);
  }
}
